// TorsionAwareTriangleUpdate_80169859547198
// MI455X (gfx1250) — compile-verified, matrix operand order checked by reading, not run on hardware
//
#include <hip/hip_runtime.h>

// ---------------------------------------------------------------------------
// TorsionAwareTriangleUpdate for MI455X (gfx1250, wave32, WMMA bf16 16x16x32)
//
// Workspace layout (bytes):
//   a_t   bf16 [B=2][64][512][512]   (b,d,i,k)   :       0 .. 67108864
//   b_t   bf16 [2][64][512][512]     (b,d,j,k)   : 67108864 .. 134217728
//   g     bf16 [2][512][512][64]     (b,i,j,d)   : 134217728 .. 201326592
//   x_t   f32  [2][64][512][512]     (b,d,i,j)   : 201326592 .. 335544320
//   wfrag bf16 pre-packed WMMA B fragments       : 335544320 .. +49152
// ---------------------------------------------------------------------------

typedef __attribute__((ext_vector_type(16))) __bf16 v16bf;
typedef __attribute__((ext_vector_type(8)))  __bf16 v8bf;
typedef __attribute__((ext_vector_type(8)))  float  v8f;
typedef int v4i_ __attribute__((vector_size(16)));

#define LL 512
#define DD 64

#if __has_builtin(__builtin_amdgcn_global_load_async_to_lds_b128) && \
    __has_builtin(__builtin_amdgcn_s_wait_asynccnt)
#define TRI_ASYNC 1
#else
#define TRI_ASYNC 0
#endif

__device__ __forceinline__ float sigmoidf_(float x) {
  return 1.0f / (1.0f + __expf(-x));
}

// ISA 7.12.2 16-bit fragment K packing: VGPR pair v holds K = {2v,2v+1} (v<4)
// or {16+2(v-4), ...} (v>=4), +8 for lanes 16..31. Fragment-order staging in
// LDS means: row-major 8-elem chunk c (k = 8c..8c+7) of row r lands at
// frag[lane = ((c&1)<<4) + r][elem = (c>>1)*8 .. +8]  (16B contiguous).
__device__ __forceinline__ int kmap8(int v, int h) {
  return ((v < 4) ? (2 * v) : (16 + 2 * (v - 4))) + 8 * h;
}

__device__ __forceinline__ v8f wmma_bf16(v16bf a, v16bf b, v8f c) {
  return __builtin_amdgcn_wmma_f32_16x16x32_bf16(
      /*neg_a=*/false, a, /*neg_b=*/false, b,
      /*c_mod=*/(short)0, c, /*reuse_a=*/false, /*reuse_b=*/false);
}

__device__ __forceinline__ v8f splat8(float x) {
  v8f v;
#pragma unroll
  for (int r = 0; r < 8; ++r) v[r] = x;
  return v;
}

// ds_swizzle group-of-32, xor_mask=16, and_mask=0x1f (SWAPX16).
__device__ __forceinline__ float xor16(float x) {
  return __uint_as_float(
      __builtin_amdgcn_ds_swizzle(__float_as_uint(x), 0x401f));
}

__device__ __forceinline__ v16bf load_wfrag(const __bf16* wfrag, int mat,
                                            int nt, int kc, int lane) {
  const size_t off = ((size_t)((mat * 4 + nt) * 2 + kc)) * 512 + lane * 16;
  return *reinterpret_cast<const v16bf*>(wfrag + off);
}

#if TRI_ASYNC
// gfx1250 async global->LDS DMA, 16B per lane; tracked by ASYNCcnt.
// Signature (from clang): (v4i AS1*, v4i AS3*, imm offset, imm cpol).
__device__ __forceinline__ void async_copy16(const __bf16* g, __bf16* l) {
  __builtin_amdgcn_global_load_async_to_lds_b128(
      (__attribute__((address_space(1))) v4i_*)g,
      (__attribute__((address_space(3))) v4i_*)l, 0, 0);
}
#endif

// ---------------------------------------------------------------------------
// Kernel 1: pre-pack 6 weight matrices (64x64 f32) into WMMA B fragments.
// out[m][n] = sum_c z[m][c] * W[n][c]  =>  B[k][n] = W[n][k].
// ---------------------------------------------------------------------------
__global__ void build_wfrags(const float* __restrict__ Wl,
                             const float* __restrict__ Wlg,
                             const float* __restrict__ Wr,
                             const float* __restrict__ Wrg,
                             const float* __restrict__ Wog,
                             const float* __restrict__ Wout,
                             __bf16* __restrict__ wfrag) {
  const int id = blockIdx.x;     // 0..47 : (mat*4 + ntile)*2 + kchunk
  const int lane = threadIdx.x;  // 0..31
  const int mat = id >> 3, nt = (id >> 1) & 3, kc = id & 1;
  const float* W = (mat == 0) ? Wl  : (mat == 1) ? Wlg : (mat == 2) ? Wr
                 : (mat == 3) ? Wrg : (mat == 4) ? Wog : Wout;
  const int n = nt * 16 + (lane & 15), h = lane >> 4;
  __bf16* dst = wfrag + (size_t)id * 512 + lane * 16;
#pragma unroll
  for (int v = 0; v < 8; ++v) {
    const int k = kc * 32 + kmap8(v, h);
    dst[2 * v]     = (__bf16)W[n * 64 + k];
    dst[2 * v + 1] = (__bf16)W[n * 64 + k + 1];
  }
}

// ---------------------------------------------------------------------------
// Kernel 2: LayerNorm + 5 gated projections. 128 rows/block (8 waves x 16).
// a (torsion-scaled, /sqrt(L)) and b go transposed to [b,d,row,k];
// g = sigmoid(z Wog^T + bog) stays in natural layout (bf16).
// ---------------------------------------------------------------------------
__global__ __launch_bounds__(256) void proj_kernel(
    const float* __restrict__ pair, const float* __restrict__ torsion,
    const float* __restrict__ ln_g, const float* __restrict__ ln_b,
    const float* __restrict__ blg, const float* __restrict__ brg,
    const float* __restrict__ bog, const float* __restrict__ Wt,
    const float* __restrict__ bt, const __bf16* __restrict__ wfrag,
    __bf16* __restrict__ a_t, __bf16* __restrict__ b_t,
    __bf16* __restrict__ g_ws) {
  __shared__ union {
    __bf16 zf[8][2][32][16];  // per-wave z fragments (phase 1)
    __bf16 g_st[128][64];     // gate staging (phase 2)
  } shA;
  __shared__ __bf16 a_st[64][128];  // [d][local k]
  __shared__ __bf16 b_st[64][128];

  const int tid = threadIdx.x, wave = tid >> 5, lane = tid & 31;
  const int m = lane & 15, h = lane >> 4;
  const long long row0 = (long long)blockIdx.x * 128;  // flattened (b,i,k)
  const int b  = (int)(row0 >> 18);
  const int i  = (int)((row0 >> 9) & 511);
  const int k0 = (int)(row0 & 511);
  const long long wrow = row0 + wave * 16 + m;

  // --- LayerNorm: 2 lanes per row, 32 elements each ---
  float xv[32];
  const float4* p4 = reinterpret_cast<const float4*>(pair + wrow * 64 + h * 32);
  float s = 0.f, s2 = 0.f;
#pragma unroll
  for (int t = 0; t < 8; ++t) {
    const float4 q = p4[t];
    xv[4 * t + 0] = q.x; xv[4 * t + 1] = q.y;
    xv[4 * t + 2] = q.z; xv[4 * t + 3] = q.w;
    s  += q.x + q.y + q.z + q.w;
    s2 += q.x * q.x + q.y * q.y + q.z * q.z + q.w * q.w;
  }
  s += xor16(s);
  s2 += xor16(s2);
  const float mean = s * (1.f / 64.f);
  const float var  = s2 * (1.f / 64.f) - mean * mean;
  const float rstd = rsqrtf(var + 1e-5f);
  // normalize + emit directly in fragment order (this lane owns frag h)
#pragma unroll
  for (int c = 0; c < 4; ++c) {
    v8bf ch;
#pragma unroll
    for (int e = 0; e < 8; ++e) {
      const int idx = h * 32 + c * 8 + e;
      ch[e] = (__bf16)((xv[c * 8 + e] - mean) * rstd * ln_g[idx] + ln_b[idx]);
    }
    *reinterpret_cast<v8bf*>(&shA.zf[wave][h][((c & 1) << 4) + m][(c >> 1) * 8]) = ch;
  }
  __syncthreads();
  const v16bf fa0 = *reinterpret_cast<const v16bf*>(&shA.zf[wave][0][lane][0]);
  const v16bf fa1 = *reinterpret_cast<const v16bf*>(&shA.zf[wave][1][lane][0]);
  __syncthreads();  // zf dead -> reuse as g_st

  // torsion gate for row i (uniform in block), fold in 1/sqrt(L)
  float twacc = bt[0];
#pragma unroll
  for (int q = 0; q < 6; ++q)
    twacc += torsion[(b * LL + i) * 6 + q] * Wt[q];
  const float ascale = sigmoidf_(twacc) / (sqrtf(512.f) + 1e-8f);

  const int kloc = wave * 16;
#pragma unroll 1
  for (int nt = 0; nt < 4; ++nt) {
    const int n = nt * 16 + m;
    // left: (z Wl^T) * sigmoid(z Wlg^T + blg) * tw / sqrt(L)
    v8f p = {};
    p = wmma_bf16(fa0, load_wfrag(wfrag, 0, nt, 0, lane), p);
    p = wmma_bf16(fa1, load_wfrag(wfrag, 0, nt, 1, lane), p);
    v8f gl = splat8(blg[n]);
    gl = wmma_bf16(fa0, load_wfrag(wfrag, 1, nt, 0, lane), gl);
    gl = wmma_bf16(fa1, load_wfrag(wfrag, 1, nt, 1, lane), gl);
    v8bf av;
#pragma unroll
    for (int r = 0; r < 8; ++r)
      av[r] = (__bf16)(p[r] * sigmoidf_(gl[r]) * ascale);
    *reinterpret_cast<v8bf*>(&a_st[n][kloc + 8 * h]) = av;
    // right: (z Wr^T) * sigmoid(z Wrg^T + brg)
    v8f pr = {};
    pr = wmma_bf16(fa0, load_wfrag(wfrag, 2, nt, 0, lane), pr);
    pr = wmma_bf16(fa1, load_wfrag(wfrag, 2, nt, 1, lane), pr);
    v8f gr = splat8(brg[n]);
    gr = wmma_bf16(fa0, load_wfrag(wfrag, 3, nt, 0, lane), gr);
    gr = wmma_bf16(fa1, load_wfrag(wfrag, 3, nt, 1, lane), gr);
    v8bf bv;
#pragma unroll
    for (int r = 0; r < 8; ++r)
      bv[r] = (__bf16)(pr[r] * sigmoidf_(gr[r]));
    *reinterpret_cast<v8bf*>(&b_st[n][kloc + 8 * h]) = bv;
    // output gate
    v8f og = splat8(bog[n]);
    og = wmma_bf16(fa0, load_wfrag(wfrag, 4, nt, 0, lane), og);
    og = wmma_bf16(fa1, load_wfrag(wfrag, 4, nt, 1, lane), og);
#pragma unroll
    for (int r = 0; r < 8; ++r)
      shA.g_st[kloc + r + 8 * h][n] = (__bf16)sigmoidf_(og[r]);
  }
  __syncthreads();

  // --- coalesced writeout: a/b -> [b,d,i,k] transposed; g -> natural ---
#pragma unroll
  for (int it = 0; it < 4; ++it) {
    const int idx = tid + it * 256;  // 1024 chunks of 8 bf16
    const int d = idx >> 4, kc2 = idx & 15;
    const long long goff =
        ((long long)(b * DD + d) * LL + i) * LL + k0 + kc2 * 8;
    *reinterpret_cast<uint4*>(a_t + goff) =
        *reinterpret_cast<const uint4*>(&a_st[d][kc2 * 8]);
    *reinterpret_cast<uint4*>(b_t + goff) =
        *reinterpret_cast<const uint4*>(&b_st[d][kc2 * 8]);
  }
  {
    const __bf16* src = &shA.g_st[0][0];
    __bf16* dst = g_ws + row0 * 64;  // 128*64 contiguous
#pragma unroll
    for (int it = 0; it < 4; ++it) {
      const int idx = tid + it * 256;
      *reinterpret_cast<uint4*>(dst + idx * 8) =
          *reinterpret_cast<const uint4*>(src + idx * 8);
    }
  }
}

// ---------------------------------------------------------------------------
// Kernel 3: per (b,d) GEMM  X = A * B^T  (512x512x512), bf16->f32.
// Workgroup: 64(i) x 128(j) tile, 8 waves of 32x32, K chunks of 32.
// LDS tiles are staged in WMMA-fragment order; async global->LDS pipeline
// (ping-pong) when the gfx1250 async builtins are available.
// ---------------------------------------------------------------------------
__global__ __launch_bounds__(256) void einsum_kernel(
    const __bf16* __restrict__ a_t, const __bf16* __restrict__ b_t,
    float* __restrict__ x_t) {
#if TRI_ASYNC
  __shared__ __bf16 fA[2][4][32][16];   // [buf][subtile][lane][elem]
  __shared__ __bf16 fB[2][8][32][16];
#else
  __shared__ __bf16 fA[1][4][32][16];
  __shared__ __bf16 fB[1][8][32][16];
#endif
  const int tid = threadIdx.x, wave = tid >> 5, lane = tid & 31;
  const int h = lane >> 4, n = lane & 15;
  const int bd = blockIdx.z;                 // b*64 + d
  const int i0 = blockIdx.y * 64, j0 = blockIdx.x * 128;
  const __bf16* __restrict__ Ab = a_t + ((long long)bd << 18);
  const __bf16* __restrict__ Bb = b_t + ((long long)bd << 18);
  const int sa = (wave >> 2) * 2;            // A subtile base (i)
  const int sb = (wave & 3) * 2;             // B subtile base (j)
  v8f acc00 = {}, acc01 = {}, acc10 = {}, acc11 = {};

  // Cooperative loader coords: A = 256 16B chunks, B = 512 (2 per thread).
  const int arow = tid >> 2, ac = tid & 3;
  __bf16* const adst =
      &fA[0][arow >> 4][((ac & 1) << 4) + (arow & 15)][(ac >> 1) * 8];
  const long long agoff = ((long long)(i0 + arow) << 9) + ac * 8;
  const int brow1 = 64 + arow;
  __bf16* const bdst0 =
      &fB[0][arow >> 4][((ac & 1) << 4) + (arow & 15)][(ac >> 1) * 8];
  __bf16* const bdst1 =
      &fB[0][brow1 >> 4][((ac & 1) << 4) + (brow1 & 15)][(ac >> 1) * 8];
  const long long bgoff0 = ((long long)(j0 + arow) << 9) + ac * 8;
  const long long bgoff1 = ((long long)(j0 + brow1) << 9) + ac * 8;

#if TRI_ASYNC
  const int ABUF = 4 * 32 * 16;  // elems per A buffer
  const int BBUF = 8 * 32 * 16;
  async_copy16(Ab + agoff, adst);
  async_copy16(Bb + bgoff0, bdst0);
  async_copy16(Bb + bgoff1, bdst1);
#pragma unroll 1
  for (int kc = 0; kc < 16; ++kc) {
    __builtin_amdgcn_s_wait_asynccnt(0);
    __syncthreads();
    const int cur = kc & 1, nxt = cur ^ 1;
    if (kc + 1 < 16) {
      const long long ko = (long long)(kc + 1) * 32;
      async_copy16(Ab + agoff + ko, adst + nxt * ABUF);
      async_copy16(Bb + bgoff0 + ko, bdst0 + nxt * BBUF);
      async_copy16(Bb + bgoff1 + ko, bdst1 + nxt * BBUF);
    }
    const v16bf a0 = *reinterpret_cast<const v16bf*>(&fA[cur][sa][lane][0]);
    const v16bf a1 = *reinterpret_cast<const v16bf*>(&fA[cur][sa + 1][lane][0]);
    const v16bf b0 = *reinterpret_cast<const v16bf*>(&fB[cur][sb][lane][0]);
    const v16bf b1 = *reinterpret_cast<const v16bf*>(&fB[cur][sb + 1][lane][0]);
    acc00 = wmma_bf16(a0, b0, acc00);
    acc01 = wmma_bf16(a0, b1, acc01);
    acc10 = wmma_bf16(a1, b0, acc10);
    acc11 = wmma_bf16(a1, b1, acc11);
  }
#else
  // register double-buffered sync pipeline
  uint4 ra  = *reinterpret_cast<const uint4*>(Ab + agoff);
  uint4 rb0 = *reinterpret_cast<const uint4*>(Bb + bgoff0);
  uint4 rb1 = *reinterpret_cast<const uint4*>(Bb + bgoff1);
#pragma unroll 1
  for (int kc = 0; kc < 16; ++kc) {
    __syncthreads();
    *reinterpret_cast<uint4*>(adst)  = ra;
    *reinterpret_cast<uint4*>(bdst0) = rb0;
    *reinterpret_cast<uint4*>(bdst1) = rb1;
    __syncthreads();
    if (kc + 1 < 16) {
      const long long ko = (long long)(kc + 1) * 32;
      ra  = *reinterpret_cast<const uint4*>(Ab + agoff + ko);
      rb0 = *reinterpret_cast<const uint4*>(Bb + bgoff0 + ko);
      rb1 = *reinterpret_cast<const uint4*>(Bb + bgoff1 + ko);
    }
    const v16bf a0 = *reinterpret_cast<const v16bf*>(&fA[0][sa][lane][0]);
    const v16bf a1 = *reinterpret_cast<const v16bf*>(&fA[0][sa + 1][lane][0]);
    const v16bf b0 = *reinterpret_cast<const v16bf*>(&fB[0][sb][lane][0]);
    const v16bf b1 = *reinterpret_cast<const v16bf*>(&fB[0][sb + 1][lane][0]);
    acc00 = wmma_bf16(a0, b0, acc00);
    acc01 = wmma_bf16(a0, b1, acc01);
    acc10 = wmma_bf16(a1, b0, acc10);
    acc11 = wmma_bf16(a1, b1, acc11);
  }
#endif

  const int wi = sa * 16, wj = sb * 16;
  float* xb = x_t + ((long long)bd << 18);
#pragma unroll
  for (int r = 0; r < 8; ++r) {
    const int mi = r + 8 * h;
    const long long ro0 = (long long)(i0 + wi + mi) << 9;
    const long long ro1 = (long long)(i0 + wi + 16 + mi) << 9;
    xb[ro0 + j0 + wj + n]      = acc00[r];
    xb[ro0 + j0 + wj + 16 + n] = acc01[r];
    xb[ro1 + j0 + wj + n]      = acc10[r];
    xb[ro1 + j0 + wj + 16 + n] = acc11[r];
  }
}

// ---------------------------------------------------------------------------
// Kernel 4: LN(x) @ Wout^T + bout, gate, residual. 128 rows/block.
// ---------------------------------------------------------------------------
__global__ __launch_bounds__(256) void out_kernel(
    const float* __restrict__ pair, const float* __restrict__ x_t,
    const __bf16* __restrict__ g_ws, const __bf16* __restrict__ wfrag,
    const float* __restrict__ lno_g, const float* __restrict__ lno_b,
    const float* __restrict__ bout, float* __restrict__ out) {
  __shared__ __bf16 yf[8][2][32][16];
  const int tid = threadIdx.x, wave = tid >> 5, lane = tid & 31;
  const int m = lane & 15, h = lane >> 4;
  const long long row0 = (long long)blockIdx.x * 128;  // flattened (b,i,j)
  const int b  = (int)(row0 >> 18);
  const int i  = (int)((row0 >> 9) & 511);
  const int j0 = (int)(row0 & 511);
  const int j  = j0 + wave * 16 + m;

  // gather x over d (coalesced across lanes in j), LN in f32
  float xv[32];
  float s = 0.f, s2 = 0.f;
#pragma unroll
  for (int e = 0; e < 32; ++e) {
    const int d = h * 32 + e;
    const float x = x_t[((long long)(b * DD + d) * LL + i) * LL + j];
    xv[e] = x; s += x; s2 += x * x;
  }
  s += xor16(s);
  s2 += xor16(s2);
  const float mean = s * (1.f / 64.f);
  const float var  = s2 * (1.f / 64.f) - mean * mean;
  const float rstd = rsqrtf(var + 1e-5f);
#pragma unroll
  for (int c = 0; c < 4; ++c) {
    v8bf ch;
#pragma unroll
    for (int e = 0; e < 8; ++e) {
      const int idx = h * 32 + c * 8 + e;
      ch[e] = (__bf16)((xv[c * 8 + e] - mean) * rstd * lno_g[idx] + lno_b[idx]);
    }
    *reinterpret_cast<v8bf*>(&yf[wave][h][((c & 1) << 4) + m][(c >> 1) * 8]) = ch;
  }
  __syncthreads();
  const v16bf fa0 = *reinterpret_cast<const v16bf*>(&yf[wave][0][lane][0]);
  const v16bf fa1 = *reinterpret_cast<const v16bf*>(&yf[wave][1][lane][0]);
  const long long rbase = row0 + wave * 16;

#pragma unroll 1
  for (int nt = 0; nt < 4; ++nt) {
    const int n = nt * 16 + m;
    v8f acc = splat8(bout[n]);
    acc = wmma_bf16(fa0, load_wfrag(wfrag, 5, nt, 0, lane), acc);
    acc = wmma_bf16(fa1, load_wfrag(wfrag, 5, nt, 1, lane), acc);
#pragma unroll
    for (int r = 0; r < 8; ++r) {
      const long long off = (rbase + r + 8 * h) * 64 + n;
      out[off] = pair[off] + acc[r] * (float)g_ws[off];
    }
  }
}

// ---------------------------------------------------------------------------
extern "C" void kernel_launch(void* const* d_in, const int* in_sizes, int n_in,
                              void* d_out, int out_size, void* d_ws,
                              size_t ws_size, hipStream_t stream) {
  const float* pair    = (const float*)d_in[0];
  const float* torsion = (const float*)d_in[1];
  const float* ln_g    = (const float*)d_in[2];
  const float* ln_b    = (const float*)d_in[3];
  const float* Wl      = (const float*)d_in[4];
  const float* Wr      = (const float*)d_in[5];
  const float* Wlg     = (const float*)d_in[6];
  const float* blg     = (const float*)d_in[7];
  const float* Wrg     = (const float*)d_in[8];
  const float* brg     = (const float*)d_in[9];
  const float* Wog     = (const float*)d_in[10];
  const float* bog     = (const float*)d_in[11];
  const float* Wout    = (const float*)d_in[12];
  const float* bout    = (const float*)d_in[13];
  const float* lno_g   = (const float*)d_in[14];
  const float* lno_b   = (const float*)d_in[15];
  const float* Wt      = (const float*)d_in[16];
  const float* bt      = (const float*)d_in[17];

  char* ws = (char*)d_ws;
  __bf16* a_t   = (__bf16*)(ws);
  __bf16* b_t   = (__bf16*)(ws + 67108864LL);
  __bf16* g_ws  = (__bf16*)(ws + 134217728LL);
  float*  x_t   = (float*) (ws + 201326592LL);
  __bf16* wfrag = (__bf16*)(ws + 335544320LL);

  build_wfrags<<<48, 32, 0, stream>>>(Wl, Wlg, Wr, Wrg, Wog, Wout, wfrag);
  proj_kernel<<<4096, 256, 0, stream>>>(pair, torsion, ln_g, ln_b, blg, brg,
                                        bog, Wt, bt, wfrag, a_t, b_t, g_ws);
  einsum_kernel<<<dim3(4, 8, 128), 256, 0, stream>>>(a_t, b_t, x_t);
  out_kernel<<<4096, 256, 0, stream>>>(pair, x_t, g_ws, wfrag, lno_g, lno_b,
                                       bout, (float*)d_out);
}
